// GraphSage_11596411699546
// MI455X (gfx1250) — matrix operand order repaired and checked by reading, not compiled, not run on hardware
//
#include <hip/hip_runtime.h>
#include <hip/hip_bf16.h>

// ---------------------------------------------------------------------------
// Types for WMMA (CDNA5 / gfx1250, wave32)
// ---------------------------------------------------------------------------
typedef __attribute__((ext_vector_type(16))) __bf16          v16bf;
typedef __attribute__((ext_vector_type(16))) unsigned short  v16u;
typedef __attribute__((ext_vector_type(8)))  unsigned short  v8u;
typedef __attribute__((ext_vector_type(8)))  float           v8f;

union BFrag  { v16u u; v16bf b; };
union AFragU { struct { v8u lo; v8u hi; } s; v16bf b; };

__device__ __forceinline__ unsigned short f2bf(float f) {
    // round-to-nearest-even f32 -> bf16
    unsigned int u = __float_as_uint(f);
    u += 0x7FFFu + ((u >> 16) & 1u);
    return (unsigned short)(u >> 16);
}

// ---------------------------------------------------------------------------
// Degree: deg[dst] += 1 over edges, then invert in place (rdeg = 1/max(deg,1))
// ---------------------------------------------------------------------------
__global__ void deg_kernel(const int* __restrict__ ei, float* __restrict__ deg, int E) {
    int e = blockIdx.x * blockDim.x + threadIdx.x;
    if (e < E) unsafeAtomicAdd(&deg[ei[E + e]], 1.0f);
}

__global__ void rdeg_kernel(float* __restrict__ deg, int n) {
    int i = blockIdx.x * blockDim.x + threadIdx.x;
    if (i < n) deg[i] = 1.0f / fmaxf(deg[i], 1.0f);
}

// ---------------------------------------------------------------------------
// Edge gather + scatter-add: agg[dst] += x[src]   (float4 / global fp atomics)
// One thread per (edge, group-of-4-channels); 2^LOGF4 threads cover one edge's row.
// ---------------------------------------------------------------------------
template<int LOGF4>
__global__ void gather_kernel(const float* __restrict__ xin, const int* __restrict__ ei,
                              float* __restrict__ agg, int E) {
    int tid = blockIdx.x * blockDim.x + threadIdx.x;
    int e = tid >> LOGF4;
    int q = tid & ((1 << LOGF4) - 1);
    if (e >= E) return;
    int s = ei[e];
    int d = ei[E + e];
    const float4 v = ((const float4*)xin)[((size_t)s << LOGF4) + q];
    float* dp = agg + ((((size_t)d << LOGF4) + q) << 2);
    unsafeAtomicAdd(dp + 0, v.x);
    unsafeAtomicAdd(dp + 1, v.y);
    unsafeAtomicAdd(dp + 2, v.z);
    unsafeAtomicAdd(dp + 3, v.w);
}

// ---------------------------------------------------------------------------
// Fused SAGE linear:  pre = (agg * rdeg) @ Wl + x @ Wr + bias
// bf16 WMMA (f32 accumulate), LDS-staged operands in wave32 fragment layout.
// Also accumulates per-column sum / sumsq for the following BatchNorm.
// Block: 256 threads (8 waves). Tile: 128(M) x 64(N). K step: 32.
// ---------------------------------------------------------------------------
template<int FIN, int FOUT>
__global__ __launch_bounds__(256)
void sage_gemm(const float* __restrict__ Agg, const float* __restrict__ X,
               const float* __restrict__ Wl,  const float* __restrict__ Wr,
               const float* __restrict__ bias, const float* __restrict__ rdeg,
               float* __restrict__ pre, float* __restrict__ colSum,
               float* __restrict__ colSumSq, int nRows)
{
    __shared__ __align__(32) unsigned short ldsA[128 * 32];  // [m][k]  bf16
    __shared__ __align__(32) unsigned short ldsB[64 * 32];   // [n][k]  bf16 (transposed W tile)

    const int t    = threadIdx.x;
    const int m0   = blockIdx.x * 128;
    const int n0   = blockIdx.y * 64;
    const int wave = t >> 5;     // 0..7  -> row strip [wave*16, wave*16+16)
    const int lane = t & 31;
    const int half = lane >> 4;  // K/row half select per ISA fragment layout
    const int ml   = lane & 15;

    v8f zero = {};
    v8f acc[4] = {zero, zero, zero, zero};

    // LDS fill assignments
    const int rA  = t >> 1;        // 0..127 : A row
    const int ksA = (t & 1) * 16;  // 0 / 16 : A k-chunk
    const int nB  = t & 63;        // 0..63  : B column
    const int ksB = (t >> 6) * 8;  // 0,8,16,24 : B k-chunk

    const float* Aptr[2] = { Agg, X };
    const float* Wptr[2] = { Wl,  Wr };

    for (int p = 0; p < 2; ++p) {
        const float* A = Aptr[p];
        const float* W = Wptr[p];
        for (int k0 = 0; k0 < FIN; k0 += 32) {
            // ---- stage A panel (f32 -> bf16, degree-normalized for phase 0)
            {
                int m = m0 + rA;
                const float* gp = nullptr;
                float sc = 1.0f;
                if (m < nRows) {
                    gp = A + (size_t)m * FIN + k0 + ksA;
                    if (p == 0) sc = rdeg[m];
                }
                unsigned short* dstA = &ldsA[rA * 32 + ksA];
                #pragma unroll
                for (int i = 0; i < 16; i += 4) {
                    float4 v = make_float4(0.f, 0.f, 0.f, 0.f);
                    if (gp) v = *(const float4*)(gp + i);
                    dstA[i + 0] = f2bf(v.x * sc);
                    dstA[i + 1] = f2bf(v.y * sc);
                    dstA[i + 2] = f2bf(v.z * sc);
                    dstA[i + 3] = f2bf(v.w * sc);
                }
            }
            // ---- stage B panel transposed: ldsB[n][k] = W[k0+k][n0+n]
            {
                #pragma unroll
                for (int i = 0; i < 8; ++i) {
                    float w = W[(size_t)(k0 + ksB + i) * FOUT + n0 + nB];
                    ldsB[nB * 32 + ksB + i] = f2bf(w);
                }
            }
            __syncthreads();

            // ---- build fragments and issue WMMA
            AFragU a;
            const unsigned short* ap = &ldsA[(wave * 16 + ml) * 32];
            a.s.lo = *(const v8u*)(ap + half * 16);       // K = {0..7} or {16..23}
            a.s.hi = *(const v8u*)(ap + half * 16 + 8);   // K = {8..15} or {24..31}

            #pragma unroll
            for (int j = 0; j < 4; ++j) {
                BFrag bf;
                bf.u = *(const v16u*)&ldsB[(j * 16 + ml) * 32 + half * 16];
                acc[j] = __builtin_amdgcn_wmma_f32_16x16x32_bf16(
                    false, a.b, false, bf.b, (short)0, acc[j], false, false);
            }
            __syncthreads();
        }
    }

    // ---- epilogue: bias, store pre-BN activations, fused BN column stats
    #pragma unroll
    for (int j = 0; j < 4; ++j) {
        int n = n0 + j * 16 + ml;
        float bv = bias[n];
        float s = 0.f, s2 = 0.f;
        #pragma unroll
        for (int r = 0; r < 8; ++r) {
            int row = m0 + wave * 16 + half * 8 + r;
            if (row < nRows) {
                float v = acc[j][r] + bv;
                pre[(size_t)row * FOUT + n] = v;
                s  += v;
                s2 += v * v;
            }
        }
        unsafeAtomicAdd(&colSum[n],   s);
        unsafeAtomicAdd(&colSumSq[n], s2);
    }
}

// ---------------------------------------------------------------------------
// BatchNorm finalize + apply (+ReLU)
// ---------------------------------------------------------------------------
__global__ void bn_stats(const float* __restrict__ colSum, const float* __restrict__ colSumSq,
                         const float* __restrict__ gamma, const float* __restrict__ beta,
                         float* __restrict__ scale, float* __restrict__ shift,
                         int nRows, int F) {
    int c = blockIdx.x * blockDim.x + threadIdx.x;
    if (c >= F) return;
    float invN = 1.0f / (float)nRows;
    float mu  = colSum[c] * invN;
    float var = colSumSq[c] * invN - mu * mu;
    float sc  = gamma[c] * rsqrtf(var + 1e-5f);
    scale[c] = sc;
    shift[c] = beta[c] - mu * sc;
}

__global__ void bn_relu(const float* __restrict__ pre, const float* __restrict__ scale,
                        const float* __restrict__ shift, float* __restrict__ h,
                        int total, int Fmask) {
    int i = blockIdx.x * blockDim.x + threadIdx.x;
    if (i >= total) return;
    int c = i & Fmask;
    h[i] = fmaxf(fmaf(pre[i], scale[c], shift[c]), 0.0f);
}

// ---------------------------------------------------------------------------
// Global mean pool (per graph) + MLP head
// ---------------------------------------------------------------------------
__global__ void pool_kernel(const float* __restrict__ h, const int* __restrict__ batch,
                            float* __restrict__ pooled, float* __restrict__ cnt, int n) {
    int tid = blockIdx.x * blockDim.x + threadIdx.x;
    int i = tid >> 6;
    int c = tid & 63;
    if (i >= n) return;
    int g = batch[i];
    unsafeAtomicAdd(&pooled[g * 64 + c], h[((size_t)i << 6) + c]);
    if (c == 0) unsafeAtomicAdd(&cnt[g], 1.0f);
}

__global__ void mlp_head(const float* __restrict__ pooled, const float* __restrict__ cnt,
                         const float* __restrict__ Wf1, const float* __restrict__ bf1,
                         const float* __restrict__ Wf2, const float* __restrict__ bf2,
                         const float* __restrict__ Wf3, const float* __restrict__ bf3,
                         float* __restrict__ out) {
    int g = threadIdx.x;  // 64 graphs, 64 threads
    float p[64], t1[64], t2[32];
    float inv = 1.0f / fmaxf(cnt[g], 1.0f);
    for (int c = 0; c < 64; ++c) p[c] = pooled[g * 64 + c] * inv;
    for (int j = 0; j < 64; ++j) {
        float s = bf1[j];
        for (int c = 0; c < 64; ++c) s = fmaf(p[c], Wf1[c * 64 + j], s);
        t1[j] = fmaxf(s, 0.f);
    }
    for (int j = 0; j < 32; ++j) {
        float s = bf2[j];
        for (int c = 0; c < 64; ++c) s = fmaf(t1[c], Wf2[c * 32 + j], s);
        t2[j] = fmaxf(s, 0.f);
    }
    for (int j = 0; j < 2; ++j) {
        float s = bf3[j];
        for (int c = 0; c < 32; ++c) s = fmaf(t2[c], Wf3[c * 2 + j], s);
        out[g * 2 + j] = s;
    }
}

// ---------------------------------------------------------------------------
// Host-side orchestration
// ---------------------------------------------------------------------------
extern "C" void kernel_launch(void* const* d_in, const int* in_sizes, int n_in,
                              void* d_out, int out_size, void* d_ws, size_t ws_size,
                              hipStream_t stream) {
    const float* x     = (const float*)d_in[0];
    const int*   ei    = (const int*)d_in[1];
    const int*   batch = (const int*)d_in[2];
    const float *Wl1 = (const float*)d_in[3],  *Wr1 = (const float*)d_in[4],
                *b1  = (const float*)d_in[5],  *g1  = (const float*)d_in[6],
                *be1 = (const float*)d_in[7];
    const float *Wl2 = (const float*)d_in[8],  *Wr2 = (const float*)d_in[9],
                *b2  = (const float*)d_in[10], *g2  = (const float*)d_in[11],
                *be2 = (const float*)d_in[12];
    const float *Wl3 = (const float*)d_in[13], *Wr3 = (const float*)d_in[14],
                *b3  = (const float*)d_in[15], *g3  = (const float*)d_in[16],
                *be3 = (const float*)d_in[17];
    const float *Wf1 = (const float*)d_in[18], *bf1 = (const float*)d_in[19],
                *Wf2 = (const float*)d_in[20], *bf2 = (const float*)d_in[21],
                *Wf3 = (const float*)d_in[22], *bf3 = (const float*)d_in[23];
    float* out = (float*)d_out;

    const int N = in_sizes[0] / 128;
    const int E = in_sizes[1] / 2;

    // Workspace carve (3 big N x 256 f32 buffers + small tails)
    char* w = (char*)d_ws;
    auto alloc = [&](size_t bytes) -> void* {
        void* p = (void*)w;
        w += (bytes + 255) & ~(size_t)255;
        return p;
    };
    float* agg      = (float*)alloc((size_t)N * 256 * 4);
    float* pre      = (float*)alloc((size_t)N * 256 * 4);
    float* hbuf     = (float*)alloc((size_t)N * 256 * 4);
    float* rdeg     = (float*)alloc((size_t)N * 4);
    float* colSum   = (float*)alloc(256 * 4);     // colSumSq is contiguous after
    float* colSumSq = (float*)alloc(256 * 4);
    float* scale    = (float*)alloc(256 * 4);
    float* shift    = (float*)alloc(256 * 4);
    float* pooled   = (float*)alloc(64 * 64 * 4); // cnt contiguous after
    float* cnt      = (float*)alloc(64 * 4);
    (void)n_in; (void)out_size; (void)ws_size; (void)colSumSq;

    // ---- degrees (graph is static across layers)
    hipMemsetAsync(rdeg, 0, (size_t)N * 4, stream);
    deg_kernel<<<(E + 255) / 256, 256, 0, stream>>>(ei, rdeg, E);
    rdeg_kernel<<<(N + 255) / 256, 256, 0, stream>>>(rdeg, N);

    const int mBlocks = (N + 127) / 128;

    // ---- Layer 1: 128 -> 256
    hipMemsetAsync(agg, 0, (size_t)N * 128 * 4, stream);
    hipMemsetAsync(colSum, 0, 2 * 256 * 4, stream);
    gather_kernel<5><<<(E * 32 + 255) / 256, 256, 0, stream>>>(x, ei, agg, E);
    sage_gemm<128, 256><<<dim3(mBlocks, 4), 256, 0, stream>>>(
        agg, x, Wl1, Wr1, b1, rdeg, pre, colSum, colSumSq, N);
    bn_stats<<<1, 256, 0, stream>>>(colSum, colSumSq, g1, be1, scale, shift, N, 256);
    bn_relu<<<(N * 256 + 255) / 256, 256, 0, stream>>>(pre, scale, shift, hbuf, N * 256, 255);

    // ---- Layer 2: 256 -> 128
    hipMemsetAsync(agg, 0, (size_t)N * 256 * 4, stream);
    hipMemsetAsync(colSum, 0, 2 * 256 * 4, stream);
    gather_kernel<6><<<(E * 64 + 255) / 256, 256, 0, stream>>>(hbuf, ei, agg, E);
    sage_gemm<256, 128><<<dim3(mBlocks, 2), 256, 0, stream>>>(
        agg, hbuf, Wl2, Wr2, b2, rdeg, pre, colSum, colSumSq, N);
    bn_stats<<<1, 256, 0, stream>>>(colSum, colSumSq, g2, be2, scale, shift, N, 128);
    bn_relu<<<(N * 128 + 255) / 256, 256, 0, stream>>>(pre, scale, shift, hbuf, N * 128, 127);

    // ---- Layer 3: 128 -> 64
    hipMemsetAsync(agg, 0, (size_t)N * 128 * 4, stream);
    hipMemsetAsync(colSum, 0, 2 * 256 * 4, stream);
    gather_kernel<5><<<(E * 32 + 255) / 256, 256, 0, stream>>>(hbuf, ei, agg, E);
    sage_gemm<128, 64><<<dim3(mBlocks, 1), 256, 0, stream>>>(
        agg, hbuf, Wl3, Wr3, b3, rdeg, pre, colSum, colSumSq, N);
    bn_stats<<<1, 64, 0, stream>>>(colSum, colSumSq, g3, be3, scale, shift, N, 64);
    bn_relu<<<(N * 64 + 255) / 256, 256, 0, stream>>>(pre, scale, shift, hbuf, N * 64, 63);

    // ---- pool + head
    hipMemsetAsync(pooled, 0, (64 * 64 + 64) * 4, stream);
    pool_kernel<<<(N * 64 + 255) / 256, 256, 0, stream>>>(hbuf, batch, pooled, cnt, N);
    mlp_head<<<1, 64, 0, stream>>>(pooled, cnt, Wf1, bf1, Wf2, bf2, Wf3, bf3, out);
}
